// Smolgen_74861279969437
// MI455X (gfx1250) — compile-verified, matrix operand order checked by reading, not run on hardware
//
#include <hip/hip_runtime.h>

// ---------------------------------------------------------------------------
// Smolgen-style pipeline for gfx1250 (MI455X), all GEMMs on v_wmma_f32_16x16x32_bf16
// ---------------------------------------------------------------------------

typedef __bf16 bf16_t;
typedef __attribute__((ext_vector_type(16))) __bf16 v16bf;
typedef __attribute__((ext_vector_type(8)))  __bf16 v8bf;
typedef __attribute__((ext_vector_type(8)))  float  v8f;

#define B_SZ 1024
#define SEQ  64
#define DM   1024
#define KC   32        // token-compress output width
#define HID  256
#define NH   16
#define PH   256
#define QD   4096
#define FLATK (SEQ*KC) // 2048
#define EPSV 1e-6f

// ---- WMMA fragment helpers (wave32, 16-bit A/B striping per CDNA5 ISA 7.12.2) ----
// A (16x32, row-major, leading dim ld): lane = 16*half + r holds row r,
//   K = half*8 + [0..7] in frag[0..7], K = 16 + half*8 + [0..7] in frag[8..15].
// B (32x16) is loaded from an N-major weight matrix W[N][K]: lane's "r" is the
//   output column, same striping over K => identical loader works for both.

__device__ __forceinline__ v16bf ld_frag_bf16(const bf16_t* __restrict__ tile, int ld) {
  const int lane = threadIdx.x & 31;
  const int half = lane >> 4;
  const int r    = lane & 15;
  const bf16_t* p = tile + (size_t)r * ld + half * 8;
  const v8bf a0 = *(const v8bf*)(p);
  const v8bf a1 = *(const v8bf*)(p + 16);
  v16bf f;
#pragma unroll
  for (int i = 0; i < 8; ++i) { f[i] = a0[i]; f[i + 8] = a1[i]; }
  return f;
}

__device__ __forceinline__ v16bf ld_frag_f32(const float* __restrict__ tile, int ld) {
  const int lane = threadIdx.x & 31;
  const int half = lane >> 4;
  const int r    = lane & 15;
  const float* p = tile + (size_t)r * ld + half * 8;
  v16bf f;
#pragma unroll
  for (int i = 0; i < 8; ++i) { f[i] = (bf16_t)p[i]; f[i + 8] = (bf16_t)p[16 + i]; }
  return f;
}

// C/D layout: VGPR v, lanes 0-15 -> row v, lanes 16-31 -> row v+8; col = lane&15.
__device__ __forceinline__ void st_tile_f32(float* __restrict__ out, int ld, v8f c) {
  const int lane = threadIdx.x & 31;
  const int half = lane >> 4;
  const int col  = lane & 15;
#pragma unroll
  for (int v = 0; v < 8; ++v) out[(size_t)(v + 8*half)*ld + col] = c[v];
}

__device__ __forceinline__ void st_tile_bf16(bf16_t* __restrict__ out, int ld, v8f c) {
  const int lane = threadIdx.x & 31;
  const int half = lane >> 4;
  const int col  = lane & 15;
#pragma unroll
  for (int v = 0; v < 8; ++v) out[(size_t)(v + 8*half)*ld + col] = (bf16_t)c[v];
}

__device__ __forceinline__ v8f wmma_bf16(v16bf a, v16bf b, v8f c) {
  return __builtin_amdgcn_wmma_f32_16x16x32_bf16(false, a, false, b, (short)0, c,
                                                 false, false);
}

// ---- weight f32 -> bf16 conversion ----
__global__ void k_cvt(const float* __restrict__ s, bf16_t* __restrict__ d, int n) {
  int i = blockIdx.x * blockDim.x + threadIdx.x;
  if (i < n) d[i] = (bf16_t)s[i];
}

// ---- Stage 1: token compress. (B*64)x1024 @ W_tc^T(1024x32) -> flat bf16 (B,2048)
__global__ void __launch_bounds__(256)
k_token_compress(const float* __restrict__ x, const bf16_t* __restrict__ Wtc,
                 bf16_t* __restrict__ flat) {
  const int wid   = (blockIdx.x * blockDim.x + threadIdx.x) >> 5;
  const int b     = wid >> 3;
  const int stile = (wid >> 1) & 3;   // 4 row tiles of 64 seq
  const int ntile = wid & 1;          // 2 col tiles of 32
  const float*  xa = x   + ((size_t)b * SEQ + stile * 16) * DM;
  const bf16_t* wb = Wtc + (size_t)(ntile * 16) * DM;
  v8f acc = {};
#pragma unroll 4
  for (int kb = 0; kb < DM; kb += 32)
    acc = wmma_bf16(ld_frag_f32(xa + kb, DM), ld_frag_bf16(wb + kb, DM), acc);
  bf16_t* out = flat + (size_t)b * FLATK + (stile * 16) * KC + ntile * 16;
  st_tile_bf16(out, KC, acc);
}

// ---- Stage 2: global compress + bias + RMSNorm. 1024x2048 @ W_gc^T -> g bf16 (B,256)
__global__ void __launch_bounds__(512)
k_global_compress(const bf16_t* __restrict__ flat, const bf16_t* __restrict__ Wgc,
                  const float* __restrict__ bias, const float* __restrict__ scale,
                  bf16_t* __restrict__ g) {
  __shared__ float C[16 * HID];
  __shared__ float part[16][32];
  __shared__ float rinv[16];
  const int mtile = blockIdx.x;        // 64 row blocks
  const int wv    = threadIdx.x >> 5;  // 16 waves = 16 N tiles
  const int lane  = threadIdx.x & 31;
  const bf16_t* A  = flat + (size_t)(mtile * 16) * FLATK;
  const bf16_t* Bm = Wgc  + (size_t)(wv * 16) * FLATK;
  v8f acc = {};
  for (int kb = 0; kb < FLATK; kb += 32)
    acc = wmma_bf16(ld_frag_bf16(A + kb, FLATK), ld_frag_bf16(Bm + kb, FLATK), acc);
  {
    const int half = lane >> 4;
    const int col  = lane & 15;
    const float bv = bias[wv * 16 + col];
#pragma unroll
    for (int v = 0; v < 8; ++v)
      C[(v + 8*half) * HID + wv * 16 + col] = acc[v] + bv;
  }
  __syncthreads();
  {
    const int row = threadIdx.x >> 5;
    const int seg = threadIdx.x & 31;
    float s = 0.f;
#pragma unroll
    for (int i = 0; i < 8; ++i) { float t = C[row * HID + seg * 8 + i]; s += t * t; }
    part[row][seg] = s;
  }
  __syncthreads();
  if (threadIdx.x < 16) {
    float s = 0.f;
#pragma unroll
    for (int i = 0; i < 32; ++i) s += part[threadIdx.x][i];
    rinv[threadIdx.x] = rsqrtf(s * (1.0f / HID) + EPSV);
  }
  __syncthreads();
  {
    const int row = threadIdx.x >> 5;
    const int cb  = (threadIdx.x & 31) * 8;
    const float r = rinv[row];
#pragma unroll
    for (int i = 0; i < 8; ++i) {
      int c = cb + i;
      g[(size_t)(mtile * 16 + row) * HID + c] = (bf16_t)(C[row * HID + c] * r * scale[c]);
    }
  }
}

// ---- Stage 3: per-head projection. 16 x (1024x256 @ 256x256) -> head bf16 (B,16,256)
__global__ void __launch_bounds__(256)
k_head_proj(const bf16_t* __restrict__ g, const bf16_t* __restrict__ Wh,
            bf16_t* __restrict__ head) {
  const int wid   = (blockIdx.x * blockDim.x + threadIdx.x) >> 5;
  const int n     = wid >> 10;        // head
  const int rem   = wid & 1023;
  const int mtile = rem >> 4;         // 64 row blocks
  const int ntile = rem & 15;         // 16 col tiles
  const bf16_t* A  = g  + (size_t)(mtile * 16) * HID;
  const bf16_t* Bm = Wh + ((size_t)n * PH + ntile * 16) * HID;
  v8f acc = {};
#pragma unroll
  for (int kb = 0; kb < HID; kb += 32)
    acc = wmma_bf16(ld_frag_bf16(A + kb, HID), ld_frag_bf16(Bm + kb, HID), acc);
  bf16_t* out = head + (size_t)(mtile * 16) * (NH * PH) + n * PH + ntile * 16;
  st_tile_bf16(out, NH * PH, acc);
}

// ---- Stage 4: logit projection (dominant). 16384x256 @ 256x4096 -> f32 d_out
// Each wave computes a 16x64 strip: A fragment reused across 4 WMMAs per K step.
__global__ void __launch_bounds__(256)
k_logits(const bf16_t* __restrict__ head, const bf16_t* __restrict__ Wl,
         float* __restrict__ out) {
  const int wid   = (blockIdx.x * blockDim.x + threadIdx.x) >> 5;
  const int mtile = wid >> 6;   // 1024 row blocks of (b, head)
  const int ngrp  = wid & 63;   // 64 groups of 4 N tiles
  const bf16_t* A = head + (size_t)(mtile * 16) * PH;
  v8f acc0 = {}, acc1 = {}, acc2 = {}, acc3 = {};
#pragma unroll
  for (int kb = 0; kb < PH; kb += 32) {
    v16bf a = ld_frag_bf16(A + kb, PH);
    const bf16_t* Bb = Wl + (size_t)(ngrp * 64) * PH + kb;
    acc0 = wmma_bf16(a, ld_frag_bf16(Bb + (size_t)0  * PH, PH), acc0);
    acc1 = wmma_bf16(a, ld_frag_bf16(Bb + (size_t)16 * PH, PH), acc1);
    acc2 = wmma_bf16(a, ld_frag_bf16(Bb + (size_t)32 * PH, PH), acc2);
    acc3 = wmma_bf16(a, ld_frag_bf16(Bb + (size_t)48 * PH, PH), acc3);
  }
  float* o = out + (size_t)(mtile * 16) * QD + ngrp * 64;
  st_tile_f32(o + 0,  QD, acc0);
  st_tile_f32(o + 16, QD, acc1);
  st_tile_f32(o + 32, QD, acc2);
  st_tile_f32(o + 48, QD, acc3);
}

// ---- workspace layout (bytes, 4KB-aligned offsets; total ~18.5 MB) ----
#define OFF_WTC   0u
#define OFF_WGC   65536u
#define OFF_WHD   1114112u
#define OFF_WLG   3211264u
#define OFF_FLAT  5308416u
#define OFF_G     9502720u
#define OFF_HEAD  10027008u

extern "C" void kernel_launch(void* const* d_in, const int* in_sizes, int n_in,
                              void* d_out, int out_size, void* d_ws, size_t ws_size,
                              hipStream_t stream) {
  const float* x     = (const float*)d_in[0];
  const float* W_tc  = (const float*)d_in[1];
  const float* W_gc  = (const float*)d_in[2];
  const float* b_gc  = (const float*)d_in[3];
  const float* rms_s = (const float*)d_in[4];
  const float* W_hd  = (const float*)d_in[5];
  const float* W_lg  = (const float*)d_in[6];
  float* out = (float*)d_out;

  char* ws = (char*)d_ws;
  bf16_t* Wtc_b  = (bf16_t*)(ws + OFF_WTC);
  bf16_t* Wgc_b  = (bf16_t*)(ws + OFF_WGC);
  bf16_t* Whd_b  = (bf16_t*)(ws + OFF_WHD);
  bf16_t* Wlg_b  = (bf16_t*)(ws + OFF_WLG);
  bf16_t* flat_b = (bf16_t*)(ws + OFF_FLAT);
  bf16_t* g_b    = (bf16_t*)(ws + OFF_G);
  bf16_t* head_b = (bf16_t*)(ws + OFF_HEAD);

  // weights -> bf16 (tiny; all stay resident in L2)
  k_cvt<<<(32 * 1024 + 255) / 256, 256, 0, stream>>>(W_tc, Wtc_b, 32 * 1024);
  k_cvt<<<(256 * 2048 + 255) / 256, 256, 0, stream>>>(W_gc, Wgc_b, 256 * 2048);
  k_cvt<<<(16 * 256 * 256 + 255) / 256, 256, 0, stream>>>(W_hd, Whd_b, 16 * 256 * 256);
  k_cvt<<<(4096 * 256 + 255) / 256, 256, 0, stream>>>(W_lg, Wlg_b, 4096 * 256);

  // 8192 waves: B * 4 Mtiles * 2 Ntiles
  k_token_compress<<<1024, 256, 0, stream>>>(x, Wtc_b, flat_b);
  // 64 blocks of 16 waves (one 16-row block each, fused RMSNorm)
  k_global_compress<<<64, 512, 0, stream>>>(flat_b, Wgc_b, b_gc, rms_s, g_b);
  // 16384 waves: 16 heads * 64 Mtiles * 16 Ntiles
  k_head_proj<<<2048, 256, 0, stream>>>(g_b, Whd_b, head_b);
  // 65536 waves: 1024 Mtiles * 64 Ngroups (16x64 strip each)
  k_logits<<<8192, 256, 0, stream>>>(head_b, Wlg_b, out);
}
